// GraphSageNet1_57243324121152
// MI455X (gfx1250) — matrix operand order repaired and checked by reading, not compiled, not run on hardware
//
#include <hip/hip_runtime.h>
#include <hip/hip_bf16.h>
#include <math.h>

// ---------------------------------------------------------------------------
// GraphSAGE (meanpool, residual) on gfx1250, v_wmma_f32_16x16x32_f16.
// Activations carried in f16 side buffers (stride 128) so WMMA A-fragments
// are two contiguous b128 loads; packed weights staged in LDS per block.
// fp32 master h kept for residual / L2-norm / readout precision.
// ---------------------------------------------------------------------------

typedef __attribute__((ext_vector_type(16))) _Float16 v16h;
typedef __attribute__((ext_vector_type(8)))  _Float16 v8h;
typedef __attribute__((ext_vector_type(8)))  float    v8f;

#define Hdim 108
#define HP   112   // fp32 row stride
#define FP   128   // f16  row stride (K padded to 4 chunks of 32)
#define NT   7     // output column tiles (108 -> 112)

// ---- A fragment from f16 row: ISA 16x32 layout == two contiguous b128 -----
// lane&15 = M ; kb = (lane>=16) ? 8 : 0 ; K = kc*32 + {kb..kb+7, 16+kb..23+kb}
__device__ __forceinline__ v16h a_frag_load16(const _Float16* __restrict__ Xrow,
                                              int kc, int lane) {
  const int kb = (lane & 16) ? 8 : 0;
  const _Float16* p = Xrow + kc * 32 + kb;
  v8h lo = *(const v8h*)(p);
  v8h hi = *(const v8h*)(p + 16);
  v16h a;
#pragma unroll
  for (int i = 0; i < 8; ++i) { a[i] = lo[i]; a[i + 8] = hi[i]; }
  return a;
}

// ---- B fragment from LDS: contiguous 16 halves per lane --------------------
__device__ __forceinline__ v16h b_frag_lds(const _Float16* smem,
                                           int chunk, int nt, int lane) {
  return *(const v16h*)(smem + ((((chunk * NT) + nt) * 32 + lane) << 4));
}

// ---- Pack fp32 weights [Krows x 108] into per-lane f16 B fragments ---------
__global__ void pack_weights(const float* __restrict__ W, int Krows, int rowOffset,
                             _Float16* __restrict__ Wp, int KC) {
  int idx = blockIdx.x * blockDim.x + threadIdx.x;
  int total = KC * NT * 32 * 16;
  if (idx >= total) return;
  int i    = idx & 15;
  int lane = (idx >> 4) & 31;
  int t    = idx >> 9;          // kc*NT + nt
  int nt   = t % NT;
  int kc   = t / NT;
  int n = nt * 16 + (lane & 15);
  int k = kc * 32 + ((lane & 16) ? 8 : 0) + ((i < 8) ? i : i + 8);
  float v = (k < Krows && n < Hdim) ? W[(long long)(rowOffset + k) * Hdim + n] : 0.0f;
  Wp[idx] = (_Float16)v;
}

// ---- Generic row-block WMMA GEMM: Y = act(Xf16 @ W + b) --------------------
template <int KC>
__global__ void wmma_gemm_rows(const _Float16* __restrict__ Xf, int ldx,
                               const _Float16* __restrict__ Wp,
                               const float* __restrict__ bias,
                               float* __restrict__ Yf32,      // may be null, stride HP
                               _Float16* __restrict__ Yf16,   // may be null, stride FP
                               int nrows, int do_relu) {
  __shared__ _Float16 smem[KC * NT * 512];
  constexpr int vecs = KC * NT * 512 / 8;          // uint4 copies
  for (int i = threadIdx.x; i < vecs; i += blockDim.x)
    ((uint4*)smem)[i] = ((const uint4*)Wp)[i];
  __syncthreads();

  const int lane = threadIdx.x & 31;
  const int wave = threadIdx.x >> 5;
  const int row0 = (blockIdx.x * (blockDim.x >> 5) + wave) * 16;
  if (row0 >= nrows) return;                       // wave-uniform
  const _Float16* Xrow = Xf + (long long)(row0 + (lane & 15)) * ldx;

  v16h aF[KC];
#pragma unroll
  for (int kc = 0; kc < KC; ++kc) aF[kc] = a_frag_load16(Xrow, kc, lane);

  v8f acc[NT] = {};
#pragma unroll
  for (int kc = 0; kc < KC; ++kc) {
#pragma unroll
    for (int nt = 0; nt < NT; ++nt) {
      v16h b = b_frag_lds(smem, kc, nt, lane);
      acc[nt] = __builtin_amdgcn_wmma_f32_16x16x32_f16(
          false, aF[kc], false, b, (short)0, acc[nt], false, false);
    }
  }

  // C/D layout: VGPR j -> M = j (+8 upper half), N = lane&15
  const int nloc  = lane & 15;
  const int rbase = row0 + ((lane & 16) ? 8 : 0);
#pragma unroll
  for (int nt = 0; nt < NT; ++nt) {
    int col = nt * 16 + nloc;
    float bv = (col < Hdim) ? bias[col] : 0.0f;
#pragma unroll
    for (int j = 0; j < 8; ++j) {
      int r = rbase + j;
      float v = acc[nt][j] + bv;
      if (do_relu) v = fmaxf(v, 0.0f);
      if (r < nrows && col < Hdim) {
        if (Yf32) Yf32[(long long)r * HP + col] = v;
        if (Yf16) Yf16[(long long)r * FP + col] = (_Float16)v;
      }
    }
  }
}

// ---- Node-apply: bundle=[h,c]@Wn+b; L2-normalize rows; h += relu(bundle) --
__global__ void wmma_apply(float* __restrict__ h,            // [N,HP] in/out fp32
                           _Float16* __restrict__ hF,        // [NR,FP] in/out f16
                           const _Float16* __restrict__ cF,  // [NR,FP] mean msgs
                           const _Float16* __restrict__ WnP, // 8 K-chunks
                           const float* __restrict__ bias,
                           int nrows) {
  __shared__ _Float16 smem[8 * NT * 512];
  constexpr int vecs = 8 * NT * 512 / 8;
  for (int i = threadIdx.x; i < vecs; i += blockDim.x)
    ((uint4*)smem)[i] = ((const uint4*)WnP)[i];
  __syncthreads();

  const int lane = threadIdx.x & 31;
  const int wave = threadIdx.x >> 5;
  const int row0 = (blockIdx.x * (blockDim.x >> 5) + wave) * 16;
  if (row0 >= nrows) return;
  const long long arow = row0 + (lane & 15);
  const _Float16* Hrow = hF + arow * FP;
  const _Float16* Crow = cF + arow * FP;

  v16h aH[4], aC[4];
#pragma unroll
  for (int kc = 0; kc < 4; ++kc) {
    aH[kc] = a_frag_load16(Hrow, kc, lane);
    aC[kc] = a_frag_load16(Crow, kc, lane);
  }

  v8f acc[NT] = {};
#pragma unroll
  for (int kc = 0; kc < 4; ++kc) {
#pragma unroll
    for (int nt = 0; nt < NT; ++nt) {
      v16h b = b_frag_lds(smem, kc, nt, lane);
      acc[nt] = __builtin_amdgcn_wmma_f32_16x16x32_f16(
          false, aH[kc], false, b, (short)0, acc[nt], false, false);
    }
  }
#pragma unroll
  for (int kc = 0; kc < 4; ++kc) {
#pragma unroll
    for (int nt = 0; nt < NT; ++nt) {
      v16h b = b_frag_lds(smem, 4 + kc, nt, lane);
      acc[nt] = __builtin_amdgcn_wmma_f32_16x16x32_f16(
          false, aC[kc], false, b, (short)0, acc[nt], false, false);
    }
  }

  const int nloc  = lane & 15;
  const int rbase = row0 + ((lane & 16) ? 8 : 0);
  float ss[8];
#pragma unroll
  for (int j = 0; j < 8; ++j) ss[j] = 0.0f;
#pragma unroll
  for (int nt = 0; nt < NT; ++nt) {
    int col = nt * 16 + nloc;
    float bv = (col < Hdim) ? bias[col] : 0.0f;
#pragma unroll
    for (int j = 0; j < 8; ++j) {
      float v = (col < Hdim) ? (acc[nt][j] + bv) : 0.0f;
      acc[nt][j] = v;
      ss[j] += v * v;
    }
  }
#pragma unroll
  for (int j = 0; j < 8; ++j) {           // reduce over 16 lanes sharing a row
    for (int off = 1; off < 16; off <<= 1)
      ss[j] += __shfl_xor(ss[j], off, 32);
    ss[j] = 1.0f / fmaxf(sqrtf(ss[j]), 1e-12f);   // F.normalize eps
  }
#pragma unroll
  for (int nt = 0; nt < NT; ++nt) {
    int col = nt * 16 + nloc;
#pragma unroll
    for (int j = 0; j < 8; ++j) {
      int r = rbase + j;
      if (r < nrows && col < Hdim) {
        long long i32 = (long long)r * HP + col;
        float o = h[i32] + fmaxf(acc[nt][j] * ss[j], 0.0f);  // residual + relu
        h[i32] = o;
        hF[(long long)r * FP + col] = (_Float16)o;
      }
    }
  }
}

// ---- Edge / readout helpers ------------------------------------------------
__global__ void fill_zero_f32(float* __restrict__ p, int n) {
  int i = blockIdx.x * blockDim.x + threadIdx.x;
  if (i < n) p[i] = 0.0f;
}
__global__ void fill_zero_f16(_Float16* __restrict__ p, long long n) {
  long long i = (long long)blockIdx.x * blockDim.x + threadIdx.x;
  if (i < n) p[i] = (_Float16)0.0f;
}
__global__ void cvt_f16(const float* __restrict__ X, _Float16* __restrict__ Y, long long n) {
  long long i = (long long)blockIdx.x * blockDim.x + threadIdx.x;
  if (i < n) Y[i] = (_Float16)X[i];
}
__global__ void degree_kernel(const int* __restrict__ dst, float* __restrict__ deg, int E) {
  int e = blockIdx.x * blockDim.x + threadIdx.x;
  if (e < E) atomicAdd(&deg[dst[e]], 1.0f);
}
__global__ void scatter_kernel(const _Float16* __restrict__ mF, const int* __restrict__ src,
                               const int* __restrict__ dst, float* __restrict__ agg, int E) {
  long long gid = (long long)blockIdx.x * blockDim.x + threadIdx.x;
  int e = (int)(gid >> 7), f = (int)(gid & 127);
  if (e < E && f < Hdim)
    atomicAdd(&agg[(long long)dst[e] * HP + f],
              (float)mF[(long long)src[e] * FP + f]);
}
__global__ void make_cF(const float* __restrict__ agg, const float* __restrict__ deg,
                        _Float16* __restrict__ cF, int N) {
  long long gid = (long long)blockIdx.x * blockDim.x + threadIdx.x;
  int v = (int)(gid >> 7), f = (int)(gid & 127);
  if (v < N && f < Hdim)
    cF[(long long)v * FP + f] =
        (_Float16)(agg[(long long)v * HP + f] / fmaxf(deg[v], 1.0f));
}
__global__ void graph_accum(const float* __restrict__ h, const int* __restrict__ gids,
                            float* __restrict__ out, int N) {
  long long gid = (long long)blockIdx.x * blockDim.x + threadIdx.x;
  int v = (int)(gid >> 7), f = (int)(gid & 127);
  if (v < N && f < Hdim)
    atomicAdd(&out[(long long)gids[v] * Hdim + f], h[(long long)v * HP + f]);
}
__global__ void graph_count(const int* __restrict__ gids, float* __restrict__ cnt, int N) {
  int v = blockIdx.x * blockDim.x + threadIdx.x;
  if (v < N) atomicAdd(&cnt[gids[v]], 1.0f);
}
__global__ void graph_div(float* __restrict__ out, const float* __restrict__ cnt, int G) {
  int i = blockIdx.x * blockDim.x + threadIdx.x;
  if (i < G * Hdim) out[i] /= fmaxf(cnt[i / Hdim], 1.0f);
}

// ---------------------------------------------------------------------------
extern "C" void kernel_launch(void* const* d_in, const int* in_sizes, int n_in,
                              void* d_out, int out_size, void* d_ws, size_t ws_size,
                              hipStream_t stream) {
  (void)n_in; (void)ws_size;
  const float* nodes_feat = (const float*)d_in[0];
  const int*   edges_src  = (const int*)d_in[4];
  const int*   edges_dst  = (const int*)d_in[5];
  const int*   graph_ids  = (const int*)d_in[6];
  const float* W_emb = (const float*)d_in[7];
  const float* b_emb = (const float*)d_in[8];
  const float* Wp1 = (const float*)d_in[9];
  const float* bp1 = (const float*)d_in[10];
  const float* Wn1 = (const float*)d_in[11];
  const float* bn1 = (const float*)d_in[12];
  const float* Wp2 = (const float*)d_in[13];
  const float* bp2 = (const float*)d_in[14];
  const float* Wn2 = (const float*)d_in[15];
  const float* bn2 = (const float*)d_in[16];

  const int N  = in_sizes[2];             // 50000
  const int E  = in_sizes[4];             // 800000
  const int G  = out_size / Hdim;         // 128
  const int IN = in_sizes[0] / N;         // 64
  const int NR = (N + 15) & ~15;          // rows padded to 16

  // ---- workspace layout ----
  char* ws = (char*)d_ws;
  size_t off = 0;
  float* h   = (float*)(ws + off); off += (size_t)N * HP * sizeof(float);
  float* agg = (float*)(ws + off); off += (size_t)N * HP * sizeof(float);
  float* deg = (float*)(ws + off); off += (size_t)N * sizeof(float);
  float* cnt = (float*)(ws + off); off += (size_t)G * sizeof(float);
  off = (off + 255) & ~(size_t)255;
  _Float16* xF = (_Float16*)(ws + off); off += (size_t)NR * IN * sizeof(_Float16);
  _Float16* hF = (_Float16*)(ws + off); off += (size_t)NR * FP * sizeof(_Float16);
  _Float16* mF = (_Float16*)(ws + off); off += (size_t)NR * FP * sizeof(_Float16);
  _Float16* cF = (_Float16*)(ws + off); off += (size_t)NR * FP * sizeof(_Float16);
  off = (off + 255) & ~(size_t)255;
  const int KCe = (IN + 31) / 32;         // 2
  _Float16* WembP = (_Float16*)(ws + off);
  _Float16* Wp1P  = WembP + (size_t)KCe * NT * 512;
  _Float16* Wp2P  = Wp1P  + (size_t)4 * NT * 512;
  _Float16* Wn1P  = Wp2P  + (size_t)4 * NT * 512;
  _Float16* Wn2P  = Wn1P  + (size_t)8 * NT * 512;

  auto cdiv = [](long long a, long long b) { return (int)((a + b - 1) / b); };

  // ---- pack weights into f16 B fragments ----
  pack_weights<<<cdiv((long long)KCe * NT * 512, 256), 256, 0, stream>>>(W_emb, IN, 0, WembP, KCe);
  pack_weights<<<cdiv(4LL * NT * 512, 256), 256, 0, stream>>>(Wp1, Hdim, 0, Wp1P, 4);
  pack_weights<<<cdiv(4LL * NT * 512, 256), 256, 0, stream>>>(Wp2, Hdim, 0, Wp2P, 4);
  pack_weights<<<cdiv(4LL * NT * 512, 256), 256, 0, stream>>>(Wn1, Hdim, 0,    Wn1P,                  4);
  pack_weights<<<cdiv(4LL * NT * 512, 256), 256, 0, stream>>>(Wn1, Hdim, Hdim, Wn1P + 4LL * NT * 512, 4);
  pack_weights<<<cdiv(4LL * NT * 512, 256), 256, 0, stream>>>(Wn2, Hdim, 0,    Wn2P,                  4);
  pack_weights<<<cdiv(4LL * NT * 512, 256), 256, 0, stream>>>(Wn2, Hdim, Hdim, Wn2P + 4LL * NT * 512, 4);

  // ---- zero pads that WMMA A-fragments read ----
  fill_zero_f16<<<cdiv((long long)NR * FP, 256), 256, 0, stream>>>(hF, (long long)NR * FP);
  fill_zero_f16<<<cdiv((long long)NR * FP, 256), 256, 0, stream>>>(cF, (long long)NR * FP);

  // ---- degree (shared by both layers) ----
  fill_zero_f32<<<cdiv(N, 256), 256, 0, stream>>>(deg, N);
  degree_kernel<<<cdiv(E, 256), 256, 0, stream>>>(edges_dst, deg, E);

  // ---- input features to f16 ----
  cvt_f16<<<cdiv((long long)N * IN, 256), 256, 0, stream>>>(nodes_feat, xF, (long long)N * IN);

  const int gemmBlocks = cdiv(N, 128);    // 256 threads = 8 waves * 16 rows

  // ---- embedding: h = X @ W_emb + b (fp32 master + f16 copy) ----
  wmma_gemm_rows<2><<<gemmBlocks, 256, 0, stream>>>(xF, IN, WembP, b_emb, h, hF, N, 0);

  // ---- layer 1 ----
  wmma_gemm_rows<4><<<gemmBlocks, 256, 0, stream>>>(hF, FP, Wp1P, bp1, nullptr, mF, N, 1);
  fill_zero_f32<<<cdiv((long long)N * HP, 256), 256, 0, stream>>>(agg, N * HP);
  scatter_kernel<<<cdiv((long long)E * 128, 256), 256, 0, stream>>>(mF, edges_src, edges_dst, agg, E);
  make_cF<<<cdiv((long long)N * 128, 256), 256, 0, stream>>>(agg, deg, cF, N);
  wmma_apply<<<gemmBlocks, 256, 0, stream>>>(h, hF, cF, Wn1P, bn1, N);

  // ---- layer 2 ----
  wmma_gemm_rows<4><<<gemmBlocks, 256, 0, stream>>>(hF, FP, Wp2P, bp2, nullptr, mF, N, 1);
  fill_zero_f32<<<cdiv((long long)N * HP, 256), 256, 0, stream>>>(agg, N * HP);
  scatter_kernel<<<cdiv((long long)E * 128, 256), 256, 0, stream>>>(mF, edges_src, edges_dst, agg, E);
  make_cF<<<cdiv((long long)N * 128, 256), 256, 0, stream>>>(agg, deg, cF, N);
  wmma_apply<<<gemmBlocks, 256, 0, stream>>>(h, hF, cF, Wn2P, bn2, N);

  // ---- readout: per-graph mean ----
  float* out = (float*)d_out;
  fill_zero_f32<<<cdiv((long long)G * Hdim, 256), 256, 0, stream>>>(out, G * Hdim);
  fill_zero_f32<<<cdiv(G, 256), 256, 0, stream>>>(cnt, G);
  graph_accum<<<cdiv((long long)N * 128, 256), 256, 0, stream>>>(h, graph_ids, out, N);
  graph_count<<<cdiv(N, 256), 256, 0, stream>>>(graph_ids, cnt, N);
  graph_div<<<cdiv((long long)G * Hdim, 256), 256, 0, stream>>>(out, cnt, G);
}
